// ClapAudioLayer_55336358643221
// MI455X (gfx1250) — matrix operand order repaired and checked by reading, not compiled, not run on hardware
//
#include <hip/hip_runtime.h>
#include <hip/hip_bf16.h>
#include <math.h>

// ---------------------------------------------------------------------------
// Swin block (ClapAudioLayer) for MI455X / gfx1250, wave32.
// GEMMs: v_wmma_f32_16x16x32_bf16, 64x128 block tile, 32x32 wave tile,
// double-buffered LDS staged with global_load_async_to_lds_b128 (ASYNCcnt).
// ---------------------------------------------------------------------------

typedef __attribute__((ext_vector_type(16))) __bf16 bf16x16;
typedef __attribute__((ext_vector_type(8)))  float  f32x8;

#define WSZ    8
#define SHIFT  4
#define HEADS  16
#define DH     32
#define CDIM   512
#define MLPD   2048
#define IMG    64
#define NTOK   32768          // B * H * W = 8*64*64
#define NWIN_G 512            // B * 64 windows

#if defined(__gfx1250__) && \
    __has_builtin(__builtin_amdgcn_global_load_async_to_lds_b128) && \
    __has_builtin(__builtin_amdgcn_s_wait_asynccnt)
#define HAVE_ASYNC_LDS 1
#else
#define HAVE_ASYNC_LDS 0
#endif

#if HAVE_ASYNC_LDS
// builtin prototype (from clang diagnostic): (int4 AS1*, int4 AS3*, Imm, Imm)
typedef __attribute__((vector_size(16))) int i32x4v;
typedef __attribute__((address_space(1))) i32x4v as1_i32x4;
typedef __attribute__((address_space(3))) i32x4v as3_i32x4;
// 16-byte global -> LDS async copy (per-lane), tracked by ASYNCcnt
__device__ inline void cp16_async(const __bf16* g, __bf16* s) {
    __builtin_amdgcn_global_load_async_to_lds_b128(
        (as1_i32x4*)(void*)g, (as3_i32x4*)(void*)s, 0, 0);
}
#define ASYNC_WAIT(n) __builtin_amdgcn_s_wait_asynccnt(n)
#else
__device__ inline void cp16_async(const __bf16* g, __bf16* s) {
    *(uint4*)s = *(const uint4*)g;
}
#define ASYNC_WAIT(n) ((void)0)
#endif

// ---- WMMA fragment loaders (ISA 7.12.2 layouts) ---------------------------
// A 16x32 bf16: lane half kb=0/8, elems [kb..kb+7] and [kb+16..kb+23]
__device__ inline bf16x16 frag_a(const __bf16* rowbase, int kb) {
    union { bf16x16 v; uint4 q[2]; } u;
    u.q[0] = *(const uint4*)(rowbase + kb);
    u.q[1] = *(const uint4*)(rowbase + kb + 16);
    return u.v;
}
// B 32x16 bf16: lane half kb2=0/16 -> kb=kb2/2, elems [kb..kb+7] and [kb+16..kb+23]
__device__ inline bf16x16 frag_b(const __bf16* rowbase, int kb2) {
    union { bf16x16 v; uint4 q[2]; } u;
    u.q[0] = *(const uint4*)(rowbase + (kb2 >> 1));
    u.q[1] = *(const uint4*)(rowbase + (kb2 >> 1) + 16);
    return u.v;
}

// ---------------------------------------------------------------------------
// One-time weight convert+transpose: Wt[n][k] (bf16) = W[k][n] (f32)
// ---------------------------------------------------------------------------
__global__ __launch_bounds__(256) void wtrans_kernel(
    const float* __restrict__ W, __bf16* __restrict__ Wt, int K, int N)
{
    unsigned idx = blockIdx.x * 256u + threadIdx.x;   // over N*K outputs
    unsigned n = idx / (unsigned)K, k = idx % (unsigned)K;
    Wt[idx] = (__bf16)W[(size_t)k * N + n];
}

// ---------------------------------------------------------------------------
// LayerNorm (one wave per token).  SHIFTED=1: fuse cyclic shift + window
// partition (output rows window-major).  SHIFTED=0: identity mapping.
// ---------------------------------------------------------------------------
template<int SHIFTED>
__global__ __launch_bounds__(256) void ln_kernel(
    const float* __restrict__ X, const float* __restrict__ g,
    const float* __restrict__ b, __bf16* __restrict__ Out)
{
    const int tok  = blockIdx.x * 8 + (threadIdx.x >> 5);
    const int lane = threadIdx.x & 31;

    size_t src;
    if (SHIFTED) {
        int bi = tok >> 12, rem = tok & 4095;
        int wi = rem >> 6, n = rem & 63;
        int wh = wi >> 3, ww = wi & 7, ii = n >> 3, jj = n & 7;
        int hs = wh * WSZ + ii, ws = ww * WSZ + jj;
        src = (size_t)bi * 4096 + (size_t)((hs + SHIFT) & 63) * IMG + ((ws + SHIFT) & 63);
    } else {
        src = (size_t)tok;
    }
    const float* x = X + src * CDIM;

    float4 v[4];
    float s = 0.f, ss = 0.f;
#pragma unroll
    for (int w = 0; w < 4; ++w) {
        v[w] = *(const float4*)(x + w * 128 + lane * 4);
        s  += v[w].x + v[w].y + v[w].z + v[w].w;
        ss += v[w].x*v[w].x + v[w].y*v[w].y + v[w].z*v[w].z + v[w].w*v[w].w;
    }
#pragma unroll
    for (int m = 16; m >= 1; m >>= 1) {
        s  += __shfl_xor(s,  m, 32);
        ss += __shfl_xor(ss, m, 32);
    }
    const float mean = s * (1.0f / CDIM);
    const float rstd = rsqrtf(ss * (1.0f / CDIM) - mean * mean + 1e-5f);

    __bf16* o = Out + (size_t)tok * CDIM;
#pragma unroll
    for (int w = 0; w < 4; ++w) {
        int c = w * 128 + lane * 4;
        o[c + 0] = (__bf16)((v[w].x - mean) * rstd * g[c + 0] + b[c + 0]);
        o[c + 1] = (__bf16)((v[w].y - mean) * rstd * g[c + 1] + b[c + 1]);
        o[c + 2] = (__bf16)((v[w].z - mean) * rstd * g[c + 2] + b[c + 2]);
        o[c + 3] = (__bf16)((v[w].w - mean) * rstd * g[c + 3] + b[c + 3]);
    }
}

// ---------------------------------------------------------------------------
// Tiled WMMA GEMM:  C[M,N] = A[M,K](bf16) * Wt[N,K](bf16, pre-transposed).
// 256 threads = 8 waves (2x4); block tile 64x128; wave tile 32x32 (2x2 frags);
// K-step 32; double-buffered LDS fed by async global->LDS copies.
// ---------------------------------------------------------------------------
enum { MODE_QKV = 0, MODE_OPROJ = 1, MODE_MLP1 = 2, MODE_MLP2 = 3 };

template<int MODE>
__global__ __launch_bounds__(256) void gemm_wmma_bf16(
    const __bf16* __restrict__ A, const __bf16* __restrict__ Wt,
    const float* __restrict__ bias, const float* __restrict__ res,
    void* __restrict__ Out, int M, int N, int K, float scale)
{
    __shared__ __bf16 sA[2][64 * 32];    // [m][k]
    __shared__ __bf16 sB[2][128 * 32];   // [n][k]

    const int blockM = blockIdx.y * 64;
    const int blockN = blockIdx.x * 128;
    const int tid  = threadIdx.x;
    const int lane = tid & 31;
    const int wave = tid >> 5;
    const int mB = (wave >> 2) * 32;     // wave tile origin in block
    const int nB = (wave & 3) * 32;
    const int kb  = (lane < 16) ? 0 : 8;
    const int kb2 = (lane < 16) ? 0 : 16;
    const int l15 = lane & 15;

    // staging assignments (pure bf16 copies, 16B chunks)
    const int rA = tid >> 2, cA = (tid & 3) * 8;      // A: 64 rows x 32
    const int rB = tid >> 1, cB = (tid & 1) * 16;     // B: 128 rows x 32

    f32x8 acc[2][2] = {};

    const int nk = K >> 5;
    auto issue = [&](int kt, int buf) {
        const int k0 = kt << 5;
        cp16_async(&A [(size_t)(blockM + rA) * K + k0 + cA], &sA[buf][rA * 32 + cA]);
        cp16_async(&Wt[(size_t)(blockN + rB) * K + k0 + cB],     &sB[buf][rB * 32 + cB]);
        cp16_async(&Wt[(size_t)(blockN + rB) * K + k0 + cB + 8], &sB[buf][rB * 32 + cB + 8]);
#if !HAVE_ASYNC_LDS
        __builtin_prefetch(&A[(size_t)(blockM + rA) * K + k0 + 32 + cA], 0, 1);
#endif
    };

    issue(0, 0);
    for (int kt = 0; kt < nk; ++kt) {
        const int buf = kt & 1;
        if (kt + 1 < nk) { issue(kt + 1, buf ^ 1); ASYNC_WAIT(3); }
        else             { ASYNC_WAIT(0); }
        __syncthreads();

        bf16x16 a0 = frag_a(&sA[buf][(mB +  0 + l15) * 32], kb);
        bf16x16 a1 = frag_a(&sA[buf][(mB + 16 + l15) * 32], kb);
        bf16x16 b0 = frag_b(&sB[buf][(nB +  0 + l15) * 32], kb2);
        bf16x16 b1 = frag_b(&sB[buf][(nB + 16 + l15) * 32], kb2);
        acc[0][0] = __builtin_amdgcn_wmma_f32_16x16x32_bf16(false, a0, false, b0, (short)0, acc[0][0], false, false);
        acc[0][1] = __builtin_amdgcn_wmma_f32_16x16x32_bf16(false, a0, false, b1, (short)0, acc[0][1], false, false);
        acc[1][0] = __builtin_amdgcn_wmma_f32_16x16x32_bf16(false, a1, false, b0, (short)0, acc[1][0], false, false);
        acc[1][1] = __builtin_amdgcn_wmma_f32_16x16x32_bf16(false, a1, false, b1, (short)0, acc[1][1], false, false);
        __syncthreads();
    }

    // ---- epilogue -----------------------------------------------------------
#pragma unroll
    for (int mt = 0; mt < 2; ++mt) {
        const int rbase = blockM + mB + mt * 16 + ((lane >= 16) ? 8 : 0);
#pragma unroll
        for (int nt = 0; nt < 2; ++nt) {
            const int colg = blockN + nB + nt * 16 + l15;
            const float bv = bias[colg];
#pragma unroll
            for (int j = 0; j < 8; ++j) {
                const int rowg = rbase + j;
                float v = acc[mt][nt][j] + bv;
                if (MODE == MODE_QKV) {
                    v *= scale;                       // folds 1/sqrt(dh) into Q
                    int wg = rowg >> 6, n = rowg & 63;
                    int h = colg >> 5, d = colg & 31;
                    ((__bf16*)Out)[(((size_t)(wg * HEADS + h)) * 64 + n) * DH + d] = (__bf16)v;
                } else if (MODE == MODE_OPROJ) {
                    int wg = rowg >> 6, n = rowg & 63;
                    int bi = wg >> 6, wi = wg & 63;
                    int wh = wi >> 3, ww = wi & 7, ii = n >> 3, jj = n & 7;
                    int hs = wh * WSZ + ii, ws = ww * WSZ + jj;
                    size_t t = (size_t)bi * 4096 + (size_t)((hs + SHIFT) & 63) * IMG
                             + ((ws + SHIFT) & 63);
                    ((float*)Out)[t * CDIM + colg] = v + res[t * CDIM + colg];
                } else if (MODE == MODE_MLP1) {
                    float gl = 0.5f * v * (1.0f + erff(v * 0.70710678118654752f));
                    ((__bf16*)Out)[(size_t)rowg * N + colg] = (__bf16)gl;
                } else { // MODE_MLP2
                    size_t o = (size_t)rowg * N + colg;
                    ((float*)Out)[o] = v + res[o];
                }
            }
        }
    }
}

// ---------------------------------------------------------------------------
// Windowed attention: 1 block per (window, head); 128 threads = 4 waves.
// scores = Q*K^T (WMMA) + rel-pos bias + shift mask; softmax; ctx = P*V (WMMA).
// ---------------------------------------------------------------------------
__global__ __launch_bounds__(128) void attn_kernel(
    const __bf16* __restrict__ Q, const __bf16* __restrict__ Km,
    const __bf16* __restrict__ V, const float* __restrict__ relTab,
    __bf16* __restrict__ CtxOut)
{
    __shared__ __bf16 sQ[64 * 32];   // [n][d]
    __shared__ __bf16 sK[64 * 32];   // [n][d]
    __shared__ __bf16 sVt[32 * 64];  // transposed: [d][n]
    __shared__ __bf16 sP[64 * 64];   // probs [m][n]

    const int wgh  = blockIdx.x;          // wg*16 + head
    const int head = wgh & 15;
    const int wg   = wgh >> 4;
    const int wi   = wg & 63;
    const int wh   = wi >> 3, ww = wi & 7;
    const size_t base = (size_t)wgh * 64 * DH;

    const int tid = threadIdx.x;
    {
        int o = tid * 16;
        *(uint4*)&sQ[o]     = *(const uint4*)&Q[base + o];
        *(uint4*)&sQ[o + 8] = *(const uint4*)&Q[base + o + 8];
        *(uint4*)&sK[o]     = *(const uint4*)&Km[base + o];
        *(uint4*)&sK[o + 8] = *(const uint4*)&Km[base + o + 8];
        union { uint4 q[2]; __bf16 h[16]; } tv;
        tv.q[0] = *(const uint4*)&V[base + o];
        tv.q[1] = *(const uint4*)&V[base + o + 8];
        int n = o >> 5, d0 = o & 31;
#pragma unroll
        for (int i = 0; i < 16; ++i) sVt[(d0 + i) * 64 + n] = tv.h[i];
    }
    __syncthreads();

    const int lane = tid & 31;
    const int wv   = tid >> 5;
    const int m0   = wv * 16;
    const int kb   = (lane < 16) ? 0 : 8;
    const int kb2  = (lane < 16) ? 0 : 16;
    const int l15  = lane & 15;
    const int mrow = m0 + l15;                       // A-fragment row
    const int myM  = m0 + ((lane >= 16) ? 8 : 0);    // D-fragment row base

    bf16x16 aq = frag_a(&sQ[mrow * DH], kb);
    float sc[4][8];

#pragma unroll
    for (int t = 0; t < 4; ++t) {
        bf16x16 bk = frag_b(&sK[(t * 16 + l15) * DH], kb2);
        f32x8 cc = {};
        cc = __builtin_amdgcn_wmma_f32_16x16x32_bf16(false, aq, false, bk,
                                                     (short)0, cc, false, false);
#pragma unroll
        for (int j = 0; j < 8; ++j) {
            int m = myM + j, n = t * 16 + l15;
            int qi = m >> 3, qj = m & 7, ki = n >> 3, kj = n & 7;
            float bias = relTab[((qi - ki + 7) * 15 + (qj - kj + 7)) * HEADS + head];
            int ph = wh * WSZ + qi, pw = ww * WSZ + qj;
            int kh = wh * WSZ + ki, kw = ww * WSZ + kj;
            int rm = (ph < 56 ? 0 : (ph < 60 ? 1 : 2)) * 3
                   + (pw < 56 ? 0 : (pw < 60 ? 1 : 2));
            int rn = (kh < 56 ? 0 : (kh < 60 ? 1 : 2)) * 3
                   + (kw < 56 ? 0 : (kw < 60 ? 1 : 2));
            sc[t][j] = cc[j] + bias + ((rm != rn) ? -100.0f : 0.0f);
        }
    }

    // row softmax over 64 keys (4 N-tiles x 16-lane half-wave reduction)
#pragma unroll
    for (int j = 0; j < 8; ++j) {
        float mx = fmaxf(fmaxf(sc[0][j], sc[1][j]), fmaxf(sc[2][j], sc[3][j]));
#pragma unroll
        for (int msk = 8; msk >= 1; msk >>= 1) mx = fmaxf(mx, __shfl_xor(mx, msk, 32));
        float e[4], sum = 0.f;
#pragma unroll
        for (int t = 0; t < 4; ++t) { e[t] = __expf(sc[t][j] - mx); sum += e[t]; }
#pragma unroll
        for (int msk = 8; msk >= 1; msk >>= 1) sum += __shfl_xor(sum, msk, 32);
        float inv = 1.0f / sum;
        int m = myM + j;
#pragma unroll
        for (int t = 0; t < 4; ++t) sP[m * 64 + t * 16 + l15] = (__bf16)(e[t] * inv);
    }
    __syncthreads();

    // ctx = P(64x64) * V(64x32)
#pragma unroll
    for (int t2 = 0; t2 < 2; ++t2) {
        f32x8 cv = {};
#pragma unroll
        for (int ks = 0; ks < 2; ++ks) {
            int k0 = ks * 32;
            bf16x16 ap = frag_a(&sP[mrow * 64 + k0], kb);
            bf16x16 bv = frag_b(&sVt[(t2 * 16 + l15) * 64 + k0], kb2);
            cv = __builtin_amdgcn_wmma_f32_16x16x32_bf16(false, ap, false, bv,
                                                         (short)0, cv, false, false);
        }
#pragma unroll
        for (int j = 0; j < 8; ++j) {
            int m = myM + j, d = t2 * 16 + l15;
            CtxOut[((size_t)wg * 64 + m) * CDIM + head * DH + d] = (__bf16)cv[j];
        }
    }
}

// ---------------------------------------------------------------------------
extern "C" void kernel_launch(void* const* d_in, const int* in_sizes, int n_in,
                              void* d_out, int out_size, void* d_ws, size_t ws_size,
                              hipStream_t stream) {
    (void)in_sizes; (void)n_in; (void)out_size; (void)ws_size;
    const float* hid_in = (const float*)d_in[0];
    const float* ln1_g  = (const float*)d_in[1];
    const float* ln1_b  = (const float*)d_in[2];
    const float* wq = (const float*)d_in[3];  const float* bq = (const float*)d_in[4];
    const float* wk = (const float*)d_in[5];  const float* bk = (const float*)d_in[6];
    const float* wv = (const float*)d_in[7];  const float* bv = (const float*)d_in[8];
    const float* rel = (const float*)d_in[9];
    const float* wo = (const float*)d_in[10]; const float* bo = (const float*)d_in[11];
    const float* ln2_g = (const float*)d_in[12];
    const float* ln2_b = (const float*)d_in[13];
    const float* w1 = (const float*)d_in[14]; const float* b1 = (const float*)d_in[15];
    const float* w2 = (const float*)d_in[16]; const float* b2 = (const float*)d_in[17];

    char* ws = (char*)d_ws;
    const size_t MB = 1u << 20;
    __bf16* xw  = (__bf16*)(ws);              // 32 MB  (aliased by ctx later)
    __bf16* qb  = (__bf16*)(ws +  32 * MB);   // 32 MB
    __bf16* kbf = (__bf16*)(ws +  64 * MB);   // 32 MB
    __bf16* vbf = (__bf16*)(ws +  96 * MB);   // 32 MB
    __bf16* ctx = (__bf16*)(ws);              // reuse xw region
    float*  hid = (float*) (ws +  32 * MB);   // 64 MB (reuse q/k regions)
    __bf16* x2  = (__bf16*)(ws +  96 * MB);   // 32 MB (reuse v region)
    __bf16* y   = (__bf16*)(ws + 128 * MB);   // 128 MB
    // pre-transposed bf16 weights [N][K]
    __bf16* wqT = (__bf16*)(ws + 256 * MB);               // 512 KB each
    __bf16* wkT = (__bf16*)(ws + 256 * MB + 512 * 1024);
    __bf16* wvT = (__bf16*)(ws + 257 * MB);
    __bf16* woT = (__bf16*)(ws + 257 * MB + 512 * 1024);
    __bf16* w1T = (__bf16*)(ws + 258 * MB);               // 2 MB
    __bf16* w2T = (__bf16*)(ws + 260 * MB);               // 2 MB

    const float qscale = 0.17677669529663687f; // 1/sqrt(32)

    // weight convert+transpose (bf16 [N][K])
    wtrans_kernel<<<CDIM * CDIM / 256, 256, 0, stream>>>(wq, wqT, CDIM, CDIM);
    wtrans_kernel<<<CDIM * CDIM / 256, 256, 0, stream>>>(wk, wkT, CDIM, CDIM);
    wtrans_kernel<<<CDIM * CDIM / 256, 256, 0, stream>>>(wv, wvT, CDIM, CDIM);
    wtrans_kernel<<<CDIM * CDIM / 256, 256, 0, stream>>>(wo, woT, CDIM, CDIM);
    wtrans_kernel<<<CDIM * MLPD / 256, 256, 0, stream>>>(w1, w1T, CDIM, MLPD);
    wtrans_kernel<<<MLPD * CDIM / 256, 256, 0, stream>>>(w2, w2T, MLPD, CDIM);

    // LN1 + shift + window partition
    ln_kernel<1><<<NTOK / 8, 256, 0, stream>>>(hid_in, ln1_g, ln1_b, xw);
    // Q,K,V projections
    gemm_wmma_bf16<MODE_QKV><<<dim3(CDIM / 128, NTOK / 64), 256, 0, stream>>>(
        xw, wqT, bq, nullptr, (void*)qb, NTOK, CDIM, CDIM, qscale);
    gemm_wmma_bf16<MODE_QKV><<<dim3(CDIM / 128, NTOK / 64), 256, 0, stream>>>(
        xw, wkT, bk, nullptr, (void*)kbf, NTOK, CDIM, CDIM, 1.0f);
    gemm_wmma_bf16<MODE_QKV><<<dim3(CDIM / 128, NTOK / 64), 256, 0, stream>>>(
        xw, wvT, bv, nullptr, (void*)vbf, NTOK, CDIM, CDIM, 1.0f);
    // attention per (window, head)
    attn_kernel<<<NWIN_G * HEADS, 128, 0, stream>>>(qb, kbf, vbf, rel, ctx);
    // output projection + window reverse + unshift + residual
    gemm_wmma_bf16<MODE_OPROJ><<<dim3(CDIM / 128, NTOK / 64), 256, 0, stream>>>(
        ctx, woT, bo, hid_in, (void*)hid, NTOK, CDIM, CDIM, 1.0f);
    // LN2
    ln_kernel<0><<<NTOK / 8, 256, 0, stream>>>(hid, ln2_g, ln2_b, x2);
    // MLP
    gemm_wmma_bf16<MODE_MLP1><<<dim3(MLPD / 128, NTOK / 64), 256, 0, stream>>>(
        x2, w1T, b1, nullptr, (void*)y, NTOK, MLPD, CDIM, 1.0f);
    gemm_wmma_bf16<MODE_MLP2><<<dim3(CDIM / 128, NTOK / 64), 256, 0, stream>>>(
        y, w2T, b2, hid, d_out, NTOK, CDIM, MLPD, 1.0f);
}
